// Quantize_18313740550489
// MI455X (gfx1250) — matrix operand order repaired and checked by reading, not compiled, not run on hardware
//
#include <hip/hip_runtime.h>

typedef __attribute__((ext_vector_type(16))) _Float16 v16h;
typedef __attribute__((ext_vector_type(8)))  float    v8f;

#define VQ_SIZE   1024
#define CDIM      64
#define GAMMA_    0.99f
#define EPS_      1e-07f

#define B_        32
#define SPATIAL   1024              /* 32 * 32 */
#define NTOK      (B_ * SPATIAL)    /* 32768 */

/* d_out layout (floats, reference tuple order) */
#define QUANT_OFF ((size_t)0)
#define ST_OFF    ((size_t)NTOK * CDIM)                  /* 2097152 */
#define IDX_OFF   ((size_t)2 * NTOK * CDIM)              /* 4194304 */
#define N_OFF     (IDX_OFF + NTOK)                       /* 4227072 */
#define ZAVG_OFF  (N_OFF + VQ_SIZE)                      /* 4228096 */
#define WNEW_OFF  (ZAVG_OFF + (size_t)VQ_SIZE * CDIM)    /* 4293632 */

/* workspace layout (bytes) */
#define WS_WHALF  0          /* 1024*64 f16  = 131072 B */
#define WS_WNORM  131072     /* 1024 f32     =   4096 B */
#define WS_CNT    135168     /* 1024 f32     =   4096 B */
#define WS_ESUM   139264     /* 1024*64 f32  = 262144 B */

/* ---- prep 1: codebook fp32 -> f16, zero encode_sum accumulator ---- */
__global__ __launch_bounds__(256) void vq_prep1(const float* __restrict__ w,
                                                _Float16* __restrict__ whalf,
                                                float* __restrict__ esum) {
    int i = blockIdx.x * 256 + threadIdx.x;        /* 0 .. 65535 */
    whalf[i] = (_Float16)w[i];
    esum[i]  = 0.0f;
}

/* ---- prep 2: per-code ||w||^2, zero counts ---- */
__global__ __launch_bounds__(256) void vq_prep2(const float* __restrict__ w,
                                                float* __restrict__ wnorm,
                                                float* __restrict__ counts) {
    int i = blockIdx.x * 256 + threadIdx.x;        /* 0 .. 1023 */
    float s = 0.0f;
    #pragma unroll 8
    for (int k = 0; k < CDIM; ++k) {
        float v = w[(size_t)i * CDIM + k];
        s += v * v;
    }
    wnorm[i]  = s;
    counts[i] = 0.0f;
}

/* ---- main: WMMA distances + argmin + quantize + scatter stats ---- */
__global__ __launch_bounds__(256) void vq_argmin_kernel(
        const float* __restrict__ z, const float* __restrict__ weight,
        const _Float16* __restrict__ whalf, const float* __restrict__ wnorm,
        float* __restrict__ counts, float* __restrict__ esum,
        float* __restrict__ out) {
    __shared__ float sA[16][CDIM + 1];    /* +1 pad: kill 16-way bank conflicts */
    __shared__ float sBest[16][128];
    __shared__ int   sIdx[16][128];
    __shared__ int   sSel[16];

    const int tid  = threadIdx.x;
    const int tok0 = blockIdx.x * 16;              /* 16 tokens per block */
    const int b    = tok0 >> 10;                   /* SPATIAL == 1024 */
    const int sp0  = tok0 & (SPATIAL - 1);

    /* stage 16x64 token tile (flat[n][ch] = z[b][ch][y][x]); 64B segments */
    {
        const int m  = tid & 15;
        const int c0 = (tid >> 4) * 4;
        const float* zb = z + (size_t)b * (CDIM * SPATIAL) + sp0 + m;
        #pragma unroll
        for (int j = 0; j < 4; ++j)
            sA[m][c0 + j] = zb[(size_t)(c0 + j) * SPATIAL];
    }
    __syncthreads();

    const int lane = tid & 31;
    const int wv   = tid >> 5;                     /* wave 0..7 */
    const int row  = lane & 15;
    const int hi   = lane >> 4;
    const int col  = lane & 15;

    /* A fragments, contiguous K order matching B: elem i -> K=hi*16+i;
       a1 covers K=32..63 */
    v16h a0, a1;
    #pragma unroll
    for (int i = 0; i < 8; ++i) {
        a0[i]     = (_Float16)sA[row][hi * 16 + i];
        a0[i + 8] = (_Float16)sA[row][hi * 16 + 8 + i];
        a1[i]     = (_Float16)sA[row][32 + hi * 16 + i];
        a1[i + 8] = (_Float16)sA[row][32 + hi * 16 + 8 + i];
    }

    float best[8];
    int   bidx[8];
    #pragma unroll
    for (int r = 0; r < 8; ++r) { best[r] = 3.4e38f; bidx[r] = 0; }

    /* each wave sweeps 128 codes; B layout: lane=N col, elem i -> K=hi*16+i */
    const int codeWaveBase = wv * 128;
    #pragma unroll
    for (int nt = 0; nt < 8; ++nt) {
        const int code = codeWaveBase + nt * 16 + col;
        const v16h b0 = *(const v16h*)(whalf + (size_t)code * CDIM + hi * 16);
        const v16h b1 = *(const v16h*)(whalf + (size_t)code * CDIM + 32 + hi * 16);
        v8f c = {};
        c = __builtin_amdgcn_wmma_f32_16x16x32_f16(false, a0, false, b0,
                                                   (short)0, c, false, false);
        c = __builtin_amdgcn_wmma_f32_16x16x32_f16(false, a1, false, b1,
                                                   (short)0, c, false, false);
        const float wn = wnorm[code];
        #pragma unroll
        for (int r = 0; r < 8; ++r) {              /* dist ~ ||w||^2 - 2 x.w */
            const float v = wn - 2.0f * c[r];
            if (v < best[r]) { best[r] = v; bidx[r] = code; }
        }
    }

    /* D layout: VGPR r holds token row = 8*hi + r */
    #pragma unroll
    for (int r = 0; r < 8; ++r) {
        const int m = hi * 8 + r;
        sBest[m][wv * 16 + col] = best[r];
        sIdx [m][wv * 16 + col] = bidx[r];
    }
    __syncthreads();

    if (tid < 16) {
        float bv = sBest[tid][0];
        int   bi = sIdx[tid][0];
        for (int j = 1; j < 128; ++j) {
            const float v = sBest[tid][j];
            const int   i = sIdx[tid][j];
            if (v < bv || (v == bv && i < bi)) { bv = v; bi = i; }
        }
        sSel[tid] = bi;
        out[IDX_OFF + tok0 + tid] = (float)bi;     /* encoding_indices */
        atomicAdd(&counts[bi], 1.0f);
    }
    __syncthreads();

    /* quantized + straight_through (numerically identical) + encode_sum */
    {
        const int m   = tid & 15;
        const int c0  = (tid >> 4) * 4;
        const int sel = sSel[m];
        const size_t base = (size_t)b * (CDIM * SPATIAL) + sp0 + m;
        #pragma unroll
        for (int j = 0; j < 4; ++j) {
            const int ch  = c0 + j;
            const float q = weight[(size_t)sel * CDIM + ch];
            out[QUANT_OFF + base + (size_t)ch * SPATIAL] = q;
            out[ST_OFF    + base + (size_t)ch * SPATIAL] = q;
            atomicAdd(&esum[(size_t)sel * CDIM + ch], sA[m][ch]);
        }
    }
}

/* ---- finalize: EMA + normalize ---- */
__global__ __launch_bounds__(1024) void vq_update_kernel(
        const float* __restrict__ N, const float* __restrict__ z_avg,
        const float* __restrict__ counts, const float* __restrict__ esum,
        float* __restrict__ out) {
    __shared__ float red[1024];
    const int i = threadIdx.x;
    const float Nn = GAMMA_ * N[i] + (1.0f - GAMMA_) * counts[i];
    out[N_OFF + i] = Nn;
    red[i] = Nn;
    __syncthreads();
    for (int s = 512; s > 0; s >>= 1) {
        if (i < s) red[i] += red[i + s];
        __syncthreads();
    }
    const float n  = red[0];
    const float wt = (Nn + EPS_) / (n + VQ_SIZE * EPS_) * n;
    #pragma unroll 4
    for (int k = 0; k < CDIM; ++k) {
        const float za = GAMMA_ * z_avg[(size_t)i * CDIM + k]
                       + (1.0f - GAMMA_) * esum[(size_t)i * CDIM + k];
        out[ZAVG_OFF + (size_t)i * CDIM + k] = za;
        out[WNEW_OFF + (size_t)i * CDIM + k] = za / wt;
    }
}

extern "C" void kernel_launch(void* const* d_in, const int* in_sizes, int n_in,
                              void* d_out, int out_size, void* d_ws, size_t ws_size,
                              hipStream_t stream) {
    const float* z      = (const float*)d_in[0];
    const float* weight = (const float*)d_in[1];
    const float* N      = (const float*)d_in[2];
    const float* z_avg  = (const float*)d_in[3];
    float* out = (float*)d_out;
    char*  ws  = (char*)d_ws;

    _Float16* whalf = (_Float16*)(ws + WS_WHALF);
    float*    wnorm = (float*)(ws + WS_WNORM);
    float*    cnt   = (float*)(ws + WS_CNT);
    float*    esum  = (float*)(ws + WS_ESUM);

    vq_prep1<<<256, 256, 0, stream>>>(weight, whalf, esum);
    vq_prep2<<<4, 256, 0, stream>>>(weight, wnorm, cnt);
    vq_argmin_kernel<<<NTOK / 16, 256, 0, stream>>>(z, weight, whalf, wnorm,
                                                    cnt, esum, out);
    vq_update_kernel<<<1, 1024, 0, stream>>>(N, z_avg, cnt, esum, out);
}
